// DeepMemoryMachine_21784074125837
// MI455X (gfx1250) — matrix operand order repaired and checked by reading, not compiled, not run on hardware
//
#include <hip/hip_runtime.h>
#include <hip/hip_bf16.h>

// ---------------------------------------------------------------------------
// DeepMemoryMachine for MI455X (gfx1250, wave32)
//   Phase A: GI = X @ W_ih^T + b_ih          (bf16 WMMA GEMM, parallel in T)
//   Phase B: sequential GRU + discrete memory (persistent kernel, LDS-resident
//            W_hh/C_w slices, grid barrier via agent-scope atomics + s_sleep)
//   Phase C: Y = H @ V_w^T + V_b             (bf16 WMMA GEMM, parallel in T)
// ---------------------------------------------------------------------------

#define T_STEPS 16384
#define N_DIM   256
#define M_DIM   512
#define G3      1536          // 3*M_DIM
#define KP1     512           // K_MEM+1
#define L_OUT   256
#define NWG     32            // persistent workgroups for the recurrence

typedef __attribute__((ext_vector_type(16))) __bf16 v16bf;
typedef __attribute__((ext_vector_type(8)))  float  v8f;

union BfPack { v16bf v; uint4 q[2]; };

// ---------------- f32 -> bf16 (round-to-nearest-even) ----------------------
__global__ __launch_bounds__(256) void f32_to_bf16_kernel(
    const float* __restrict__ src, unsigned short* __restrict__ dst, int n) {
  int i = blockIdx.x * 256 + threadIdx.x;
  if (i < n) {
    unsigned u = __float_as_uint(src[i]);
    u += 0x7FFFu + ((u >> 16) & 1u);
    dst[i] = (unsigned short)(u >> 16);
  }
}

// ---------------- state init (ws is poisoned by harness) -------------------
__global__ __launch_bounds__(256) void init_state_kernel(
    float* __restrict__ Mem, int* __restrict__ written,
    float* __restrict__ h_vec, const float* __restrict__ h0,
    int* __restrict__ bar_cnt, float* __restrict__ cand_val,
    int* __restrict__ cand_idx) {
  int i = blockIdx.x * 256 + threadIdx.x;
  if (i < KP1 * M_DIM) Mem[i] = 0.0f;
  if (i < KP1) written[i] = 0;
  if (i < M_DIM) h_vec[i] = h0[i];
  if (i < NWG) { cand_val[i] = 0.0f; cand_idx[i] = 0; }
  if (i == 0) *bar_cnt = 0;
}

// ---------------- Phase A: GI[t, 3m] = X @ W_ih^T + b_ih -------------------
// one wave per 16x16 output tile, K = 256 in 8 WMMA steps
__global__ __launch_bounds__(256) void gi_gemm_kernel(
    const unsigned short* __restrict__ Xb,   // [T, 256] bf16
    const unsigned short* __restrict__ Wb,   // [1536, 256] bf16
    const float* __restrict__ b_ih,          // [1536]
    float* __restrict__ GI) {                // [T, 1536]
  const int wid = threadIdx.x >> 5;
  const int l   = threadIdx.x & 31;
  const int kl  = l & 15;
  const int sel = l >> 4;
  const int tile = blockIdx.x * 8 + wid;
  const int tm = tile / (G3 / 16);           // T tile
  const int tn = tile % (G3 / 16);           // 3m tile
  const unsigned short* arow = Xb + (size_t)(tm * 16 + kl) * N_DIM;
  const unsigned short* brow = Wb + (size_t)(tn * 16 + kl) * N_DIM;
  const float bias = b_ih[tn * 16 + kl];
  v8f c;
#pragma unroll
  for (int r = 0; r < 8; ++r) c[r] = bias;
#pragma unroll
  for (int kb = 0; kb < N_DIM; kb += 32) {
    BfPack a, b;
    // A 16x32 bf16 per-lane layout: K = sel*8..+7 and K = 16+sel*8..+7
    a.q[0] = *reinterpret_cast<const uint4*>(arow + kb + sel * 8);
    a.q[1] = *reinterpret_cast<const uint4*>(arow + kb + 16 + sel * 8);
    // B 32x16 bf16 per-lane layout: N = kl, K = sel*8..+7 and K = 16+sel*8..+7
    b.q[0] = *reinterpret_cast<const uint4*>(brow + kb + sel * 8);
    b.q[1] = *reinterpret_cast<const uint4*>(brow + kb + 16 + sel * 8);
    c = __builtin_amdgcn_wmma_f32_16x16x32_bf16(false, a.v, false, b.v,
                                                (short)0, c, false, false);
  }
  float* orow = GI + (size_t)(tm * 16 + sel * 8) * G3 + tn * 16 + kl;
#pragma unroll
  for (int r = 0; r < 8; ++r) orow[(size_t)r * G3] = c[r];
}

// ---------------- Phase C: Y = H @ V_w^T + V_b -----------------------------
__global__ __launch_bounds__(256) void y_gemm_kernel(
    const unsigned short* __restrict__ Hb,   // [T, 512] bf16
    const unsigned short* __restrict__ Vb,   // [256, 512] bf16
    const float* __restrict__ V_b,           // [256]
    float* __restrict__ Y) {                 // [T, 256]
  const int wid = threadIdx.x >> 5;
  const int l   = threadIdx.x & 31;
  const int kl  = l & 15;
  const int sel = l >> 4;
  const int tile = blockIdx.x * 8 + wid;
  const int tm = tile / (L_OUT / 16);
  const int tn = tile % (L_OUT / 16);
  const unsigned short* arow = Hb + (size_t)(tm * 16 + kl) * M_DIM;
  const unsigned short* brow = Vb + (size_t)(tn * 16 + kl) * M_DIM;
  const float bias = V_b[tn * 16 + kl];
  v8f c;
#pragma unroll
  for (int r = 0; r < 8; ++r) c[r] = bias;
#pragma unroll
  for (int kb = 0; kb < M_DIM; kb += 32) {
    BfPack a, b;
    a.q[0] = *reinterpret_cast<const uint4*>(arow + kb + sel * 8);
    a.q[1] = *reinterpret_cast<const uint4*>(arow + kb + 16 + sel * 8);
    b.q[0] = *reinterpret_cast<const uint4*>(brow + kb + sel * 8);
    b.q[1] = *reinterpret_cast<const uint4*>(brow + kb + 16 + sel * 8);
    c = __builtin_amdgcn_wmma_f32_16x16x32_bf16(false, a.v, false, b.v,
                                                (short)0, c, false, false);
  }
  float* orow = Y + (size_t)(tm * 16 + sel * 8) * L_OUT + tn * 16 + kl;
#pragma unroll
  for (int r = 0; r < 8; ++r) orow[(size_t)r * L_OUT] = c[r];
}

// ---------------- grid-wide barrier (monotonic counter) --------------------
__device__ __forceinline__ void grid_sync(int* cnt, int target) {
  __threadfence();
  __syncthreads();
  if (threadIdx.x == 0) {
    __hip_atomic_fetch_add(cnt, 1, __ATOMIC_ACQ_REL, __HIP_MEMORY_SCOPE_AGENT);
    while (__hip_atomic_load(cnt, __ATOMIC_ACQUIRE,
                             __HIP_MEMORY_SCOPE_AGENT) < target) {
      __builtin_amdgcn_s_sleep(2);
    }
  }
  __syncthreads();
}

// ---------------- Phase B: sequential GRU + discrete memory ----------------
// 32 persistent WGs x 256 threads. WG w owns hidden indices [16w, 16w+16) and
// logit rows [16w, 16w+16). LDS: 48x512 W_hh slice + 16x512 C_w slice + h buf.
__global__ __launch_bounds__(256, 1) void dmm_recurrent_kernel(
    const float* __restrict__ GI,   const float* __restrict__ W_hh,
    const float* __restrict__ b_hh, const float* __restrict__ C_w,
    const float* __restrict__ C_b,  float* __restrict__ h_vec,
    float* __restrict__ hnew_vec,   float* __restrict__ Mem,
    int* __restrict__ written,      float* __restrict__ cand_val,
    int* __restrict__ cand_idx,     int* __restrict__ bar_cnt,
    float* __restrict__ H) {
  extern __shared__ float smem[];
  float* Wlds = smem;               // 48*512 = 24576 floats
  float* Clds = smem + 24576;       // 16*512 =  8192 floats
  float* hbuf = smem + 32768;       // 512 floats
  float* rval = smem + 33280;       // 16 floats (per-WG logits)
  int*   qsh  = (int*)(smem + 33296);

  const int wg  = blockIdx.x;
  const int tid = threadIdx.x;
  const int wv  = tid >> 5;
  const int l   = tid & 31;

  // cache this WG's weight slices in LDS (loaded once, reused 16384x)
  for (int idx = tid; idx < 48 * 512; idx += 256) {
    int rloc = idx >> 9, k = idx & 511;
    int i_loc = rloc / 3, g = rloc - i_loc * 3;
    Wlds[idx] = W_hh[(size_t)(g * M_DIM + wg * 16 + i_loc) * M_DIM + k];
  }
  for (int idx = tid; idx < 16 * 512; idx += 256) {
    int jloc = idx >> 9, k = idx & 511;
    Clds[idx] = C_w[(size_t)(wg * 16 + jloc) * M_DIM + k];
  }

  int epoch = 0;
  for (int t = 0; t < T_STEPS; ++t) {
    // ---- phase 1: gh = W_hh @ h ; h_new via GRU gates ----
    for (int k = tid; k < M_DIM; k += 256) hbuf[k] = h_vec[k];
    __syncthreads();
    if (t + 1 < T_STEPS)  // warm next step's GI row into cache
      __builtin_prefetch(GI + (size_t)(t + 1) * G3 + tid * 6, 0, 0);
#pragma unroll
    for (int it = 0; it < 2; ++it) {
      const int i_loc = wv * 2 + it;        // each wave handles 2 indices
      const float* wr0 = Wlds + i_loc * 1536;
      float sr = 0.f, sz = 0.f, sn = 0.f;
#pragma unroll
      for (int kk = 0; kk < 16; ++kk) {
        int k = kk * 32 + l;                // bank-conflict-free
        float hv = hbuf[k];
        sr = fmaf(wr0[k],        hv, sr);
        sz = fmaf(wr0[512 + k],  hv, sz);
        sn = fmaf(wr0[1024 + k], hv, sn);
      }
#pragma unroll
      for (int m = 16; m; m >>= 1) {
        sr += __shfl_xor(sr, m, 32);
        sz += __shfl_xor(sz, m, 32);
        sn += __shfl_xor(sn, m, 32);
      }
      if (l == 0) {
        const int i = wg * 16 + i_loc;
        const float* gi = GI + (size_t)t * G3;
        float rr = 1.f / (1.f + __expf(-(gi[i]         + sr + b_hh[i])));
        float zz = 1.f / (1.f + __expf(-(gi[M_DIM + i] + sz + b_hh[M_DIM + i])));
        float ng = tanhf(gi[2 * M_DIM + i] + rr * (sn + b_hh[2 * M_DIM + i]));
        hnew_vec[i] = (1.f - zz) * ng + zz * hbuf[i];
      }
    }
    grid_sync(bar_cnt, (++epoch) * NWG);

    // ---- phase 2: logits = C_w @ h_new ; per-WG argmax candidate ----
    for (int k = tid; k < M_DIM; k += 256) hbuf[k] = hnew_vec[k];
    __syncthreads();
#pragma unroll
    for (int it = 0; it < 2; ++it) {
      const int j_loc = wv * 2 + it;
      const float* cr = Clds + j_loc * 512;
      float sl = 0.f;
#pragma unroll
      for (int kk = 0; kk < 16; ++kk) {
        int k = kk * 32 + l;
        sl = fmaf(cr[k], hbuf[k], sl);
      }
#pragma unroll
      for (int m = 16; m; m >>= 1) sl += __shfl_xor(sl, m, 32);
      if (l == 0) rval[j_loc] = sl + C_b[wg * 16 + j_loc];
    }
    __syncthreads();
    if (tid == 0) {  // ascending scan + strict '>' => lowest-index tie-break
      float bv = -3.4e38f; int bi = 0;
      for (int j = 0; j < 16; ++j) {
        float v = rval[j];
        if (v > bv) { bv = v; bi = wg * 16 + j; }
      }
      cand_val[wg] = bv; cand_idx[wg] = bi;
    }
    grid_sync(bar_cnt, (++epoch) * NWG);

    // ---- phase 3: global argmax + memory read/write (WG0 only) ----
    if (wg == 0) {
      if (tid == 0) {
        float bv = -3.4e38f; int q = 0;
        for (int w2 = 0; w2 < NWG; ++w2) {
          float v = cand_val[w2];
          if (v > bv) { bv = v; q = cand_idx[w2]; }
        }
        qsh[0] = q;
      }
      __syncthreads();
      const int q = qsh[0];
      const int wr = written[q];
      const bool hasq = (q > 0) && (wr != 0);   // read memory -> h_out
      const bool dow  = (q > 0) && (wr == 0);   // first write of this address
      for (int e = tid; e < M_DIM; e += 256) {
        float hn = hnew_vec[e];
        float mv = Mem[(size_t)q * M_DIM + e];
        float ho = hasq ? mv : hn;
        if (dow) Mem[(size_t)q * M_DIM + e] = hn;
        H[(size_t)t * M_DIM + e] = ho;
        h_vec[e] = ho;
      }
      if (tid == 0 && q > 0) written[q] = 1;
    }
    grid_sync(bar_cnt, (++epoch) * NWG);
  }
}

// ---------------------------------------------------------------------------
extern "C" void kernel_launch(void* const* d_in, const int* in_sizes, int n_in,
                              void* d_out, int out_size, void* d_ws,
                              size_t ws_size, hipStream_t stream) {
  const float* X    = (const float*)d_in[0];
  const float* h0   = (const float*)d_in[1];
  const float* W_ih = (const float*)d_in[2];
  const float* W_hh = (const float*)d_in[3];
  const float* b_ih = (const float*)d_in[4];
  const float* b_hh = (const float*)d_in[5];
  const float* C_w  = (const float*)d_in[6];
  const float* C_b  = (const float*)d_in[7];
  const float* V_w  = (const float*)d_in[8];
  const float* V_b  = (const float*)d_in[9];
  float* Y = (float*)d_out;

  char* ws = (char*)d_ws;
  size_t off = 0;
  auto carve = [&](size_t bytes) -> void* {
    void* p = ws + off;
    off += (bytes + 255) & ~(size_t)255;
    return p;
  };
  float*          GI   = (float*)carve((size_t)T_STEPS * G3 * 4);
  float*          H    = (float*)carve((size_t)T_STEPS * M_DIM * 4);
  unsigned short* Xb   = (unsigned short*)carve((size_t)T_STEPS * N_DIM * 2);
  unsigned short* Wb   = (unsigned short*)carve((size_t)G3 * N_DIM * 2);
  unsigned short* Hb   = (unsigned short*)carve((size_t)T_STEPS * M_DIM * 2);
  unsigned short* Vb   = (unsigned short*)carve((size_t)L_OUT * M_DIM * 2);
  float*          Mem  = (float*)carve((size_t)KP1 * M_DIM * 4);
  float*          hv   = (float*)carve(M_DIM * 4);
  float*          hnv  = (float*)carve(M_DIM * 4);
  float*          cval = (float*)carve(NWG * 4);
  int*            cidx = (int*)carve(NWG * 4);
  int*            wrt  = (int*)carve(KP1 * 4);
  int*            bar  = (int*)carve(256);

  // state init (ws is poisoned by the harness; re-zero every launch)
  init_state_kernel<<<(KP1 * M_DIM + 255) / 256, 256, 0, stream>>>(
      Mem, wrt, hv, h0, bar, cval, cidx);

  // bf16 staging for WMMA GEMMs
  f32_to_bf16_kernel<<<(T_STEPS * N_DIM + 255) / 256, 256, 0, stream>>>(
      X, Xb, T_STEPS * N_DIM);
  f32_to_bf16_kernel<<<(G3 * N_DIM + 255) / 256, 256, 0, stream>>>(
      W_ih, Wb, G3 * N_DIM);
  f32_to_bf16_kernel<<<(L_OUT * M_DIM + 255) / 256, 256, 0, stream>>>(
      V_w, Vb, L_OUT * M_DIM);

  // Phase A: GI = X @ W_ih^T + b_ih  (1024*96 tiles, 8 waves/block)
  gi_gemm_kernel<<<(T_STEPS / 16) * (G3 / 16) / 8, 256, 0, stream>>>(
      Xb, Wb, b_ih, GI);

  // Phase B: persistent recurrence (32 WGs, ~130 KB dynamic LDS each)
  const size_t smem_bytes = (size_t)33312 * 4;
  dmm_recurrent_kernel<<<NWG, 256, smem_bytes, stream>>>(
      GI, W_hh, b_hh, C_w, C_b, hv, hnv, Mem, wrt, cval, cidx, bar, H);

  // Phase C: Y = H @ V_w^T + V_b
  f32_to_bf16_kernel<<<(T_STEPS * M_DIM + 255) / 256, 256, 0, stream>>>(
      H, Hb, T_STEPS * M_DIM);
  y_gemm_kernel<<<(T_STEPS / 16) * (L_OUT / 16) / 8, 256, 0, stream>>>(
      Hb, Vb, V_b, Y);
}
